// SetAbstract_25220047962579
// MI455X (gfx1250) — compile-verified, matrix operand order checked by reading, not run on hardware
//
#include <hip/hip_runtime.h>
#include <hip/hip_bf16.h>

typedef __attribute__((ext_vector_type(16))) _Float16 v16h;
typedef __attribute__((ext_vector_type(8)))  _Float16 v8h;
typedef __attribute__((ext_vector_type(8)))  float    v8f;

#define B_   4
#define N_   8192
#define S_   2048
#define K_   16
#define NN_  (B_*S_*K_)      /* 131072 flat samples */
#define LEAKY_ 0.1f
#define BN_EPS_ 1e-5f

// ---------------------------------------------------------------------------
// Helpers
// ---------------------------------------------------------------------------
__device__ inline v16h cat8(v8h lo, v8h hi) {
    v16h r;
#pragma unroll
    for (int j = 0; j < 8; ++j) { r[j] = lo[j]; r[8 + j] = hi[j]; }
    return r;
}

// ---------------------------------------------------------------------------
// 0. zero a float buffer (BN sum/sumsq accumulators)
// ---------------------------------------------------------------------------
__global__ void zero_f32(float* p, int n) {
    int i = blockIdx.x * blockDim.x + threadIdx.x;
    if (i < n) p[i] = 0.f;
}

// ---------------------------------------------------------------------------
// 1. pack conv weights to f16, pad K to CinPad; layer0 reorders columns
//    (src cols = [3 coords][64 feats] -> dst = [64 feats][3 coords][pad])
// ---------------------------------------------------------------------------
__global__ void pack_w(const float* __restrict__ src, _Float16* __restrict__ dst,
                       int rows, int srcCols, int dstCols, int reorder) {
    int tid = blockIdx.x * blockDim.x + threadIdx.x;
    if (tid >= rows * dstCols) return;
    int r = tid / dstCols, c = tid % dstCols;
    int sc;
    if (reorder) sc = (c < 64) ? (3 + c) : ((c < 67) ? (c - 64) : -1);
    else         sc = (c < srcCols) ? c : -1;
    dst[tid] = (sc >= 0) ? (_Float16)src[r * srcCols + sc] : (_Float16)0.f;
}

// ---------------------------------------------------------------------------
// 2. transpose points [B,64,N] f32 -> ptsT [B,N,64] f16 (coalesced reads)
// ---------------------------------------------------------------------------
__global__ void pts_transpose(const float* __restrict__ points, _Float16* __restrict__ ptsT) {
    int tid = blockIdx.x * blockDim.x + threadIdx.x;   // b*64*N + c*N + n
    if (tid >= B_ * 64 * N_) return;
    int n = tid % N_;
    int c = (tid / N_) % 64;
    int b = tid / (64 * N_);
    ptsT[((size_t)b * N_ + n) * 64 + c] = (_Float16)points[tid];
}

// ---------------------------------------------------------------------------
// 3. FPS: one block per batch, 1024 threads = 32 waves. Sequential 2048
//    rounds of min-update + block argmax (shfl tree + LDS across waves).
// ---------------------------------------------------------------------------
__global__ void fps_kernel(const float* __restrict__ xyz, float* __restrict__ dist,
                           float* __restrict__ newxyz_ws,
                           float* __restrict__ out_newxyz, int* __restrict__ out_idx) {
    int b = blockIdx.x;
    const float* px = xyz + (size_t)b * 3 * N_;
    const float* py = px + N_;
    const float* pz = px + 2 * N_;
    float* dst = dist + (size_t)b * N_;
    int tid = threadIdx.x;

    for (int n = tid; n < N_; n += 1024) dst[n] = 1e10f;

    __shared__ float s_bd[32];
    __shared__ int   s_bi[32];
    __shared__ int   s_far;
    if (tid == 0) s_far = 0;
    __syncthreads();

    for (int it = 0; it < S_; ++it) {
        int far = s_far;
        float cx = px[far], cy = py[far], cz = pz[far];
        if (tid == 0) {
            out_idx[b * S_ + it] = far;
            out_newxyz[(size_t)b * 3 * S_ + it]          = cx;
            out_newxyz[(size_t)b * 3 * S_ + S_ + it]     = cy;
            out_newxyz[(size_t)b * 3 * S_ + 2 * S_ + it] = cz;
            newxyz_ws[((size_t)b * S_ + it) * 3 + 0] = cx;
            newxyz_ws[((size_t)b * S_ + it) * 3 + 1] = cy;
            newxyz_ws[((size_t)b * S_ + it) * 3 + 2] = cz;
        }
        float bd = -1.f; int bi = 0;
        for (int n = tid; n < N_; n += 1024) {
            float dx = px[n] - cx, dy = py[n] - cy, dz = pz[n] - cz;
            float d = dx * dx + dy * dy + dz * dz;
            float dn = fminf(dst[n], d);
            dst[n] = dn;
            if (dn > bd) { bd = dn; bi = n; }
        }
#pragma unroll
        for (int off = 16; off >= 1; off >>= 1) {
            float od = __shfl_xor(bd, off, 32);
            int   oi = __shfl_xor(bi, off, 32);
            if (od > bd) { bd = od; bi = oi; }
        }
        if ((tid & 31) == 0) { s_bd[tid >> 5] = bd; s_bi[tid >> 5] = bi; }
        __syncthreads();
        if (tid < 32) {
            bd = s_bd[tid]; bi = s_bi[tid];
#pragma unroll
            for (int off = 16; off >= 1; off >>= 1) {
                float od = __shfl_xor(bd, off, 32);
                int   oi = __shfl_xor(bi, off, 32);
                if (od > bd) { bd = od; bi = oi; }
            }
            if (tid == 0) s_far = bi;
        }
        __syncthreads();
    }
}

// ---------------------------------------------------------------------------
// 4. kNN top-16: one wave per query; per-lane sorted top-16 in registers,
//    then 16-round cross-lane min merge (shfl only, no LDS, no spills).
// ---------------------------------------------------------------------------
__global__ void knn_kernel(const float* __restrict__ xyz,
                           const float* __restrict__ newxyz_ws,
                           int* __restrict__ idx_ws) {
    int gw   = (blockIdx.x * blockDim.x + threadIdx.x) >> 5;   // query 0..B*S-1
    int lane = threadIdx.x & 31;
    if (gw >= B_ * S_) return;
    int b = gw / S_;
    const float* px = xyz + (size_t)b * 3 * N_;
    float qx = newxyz_ws[gw * 3 + 0];
    float qy = newxyz_ws[gw * 3 + 1];
    float qz = newxyz_ws[gw * 3 + 2];

    float dk[16]; int ik[16];
#pragma unroll
    for (int j = 0; j < 16; ++j) { dk[j] = 1e30f; ik[j] = 0; }

    for (int n = lane; n < N_; n += 32) {
        float dx = px[n] - qx, dy = px[N_ + n] - qy, dz = px[2 * N_ + n] - qz;
        float d = dx * dx + dy * dy + dz * dz;
        if (d < dk[15]) {
            float cd = d; int ci = n;
#pragma unroll
            for (int j = 0; j < 16; ++j) {
                if (cd < dk[j]) {
                    float td = dk[j]; dk[j] = cd; cd = td;
                    int   ti = ik[j]; ik[j] = ci; ci = ti;
                }
            }
        }
    }
    // merge 32 sorted lists: each round pick global min head, pop winner.
    for (int r = 0; r < 16; ++r) {
        float bd = dk[0]; int bi = ik[0]; int bl = lane;
#pragma unroll
        for (int off = 16; off >= 1; off >>= 1) {
            float od = __shfl_xor(bd, off, 32);
            int   oi = __shfl_xor(bi, off, 32);
            int   ol = __shfl_xor(bl, off, 32);
            if (od < bd || (od == bd && ol < bl)) { bd = od; bi = oi; bl = ol; }
        }
        if (lane == 0) idx_ws[gw * 16 + r] = bi;
        if (lane == bl) {
#pragma unroll
            for (int j = 0; j < 15; ++j) { dk[j] = dk[j + 1]; ik[j] = ik[j + 1]; }
            dk[15] = 1e30f;
        }
    }
}

// ---------------------------------------------------------------------------
// 5. gather + concat -> X0 [NN][96] f16 : [64 feats][3 rel coords][29 zero]
// ---------------------------------------------------------------------------
__global__ void gather_kernel(const float* __restrict__ xyz,
                              const _Float16* __restrict__ ptsT,
                              const float* __restrict__ newxyz_ws,
                              const int* __restrict__ idx_ws,
                              _Float16* __restrict__ X0) {
    int tid = blockIdx.x * blockDim.x + threadIdx.x;   // sample (b,s,k)
    if (tid >= NN_) return;
    int bs = tid >> 4;
    int b  = bs / S_;
    int n  = idx_ws[tid];
    const v8h* src = (const v8h*)(ptsT + ((size_t)b * N_ + n) * 64);
    v8h* dst = (v8h*)(X0 + (size_t)tid * 96);
#pragma unroll
    for (int j = 0; j < 8; ++j) dst[j] = src[j];
    float gx = xyz[(size_t)b * 3 * N_ + n]           - newxyz_ws[bs * 3 + 0];
    float gy = xyz[(size_t)b * 3 * N_ + N_ + n]      - newxyz_ws[bs * 3 + 1];
    float gz = xyz[(size_t)b * 3 * N_ + 2 * N_ + n]  - newxyz_ws[bs * 3 + 2];
    v8h z;
#pragma unroll
    for (int j = 0; j < 8; ++j) z[j] = (_Float16)0.f;
    v8h t = z;
    t[0] = (_Float16)gx; t[1] = (_Float16)gy; t[2] = (_Float16)gz;
    dst[8] = t; dst[9] = z; dst[10] = z; dst[11] = z;
}

// ---------------------------------------------------------------------------
// 6. WMMA GEMM + fused BN-statistics, fully templated so all strides and
//    K-trip counts are compile-time: straight-line wmma chains, immediate
//    IOFFSET b128 loads, software double-buffered B fragments across the
//    8 N-subtiles. Y[n][cout] = sum_c W[cout][c] * X[n][c].
//    Per ISA 16-bit fragment layout: lane halves 0..7 = K g*8..g*8+7,
//    halves 8..15 = K g*8+16..g*8+23  -> two b128 loads per fragment.
// ---------------------------------------------------------------------------
template <int KT, int CIN, int COUT, int MT>
__global__ void gemm_bn(const _Float16* __restrict__ X, const _Float16* __restrict__ W,
                        _Float16* __restrict__ Y,
                        float* __restrict__ sumv, float* __restrict__ sumsq) {
    constexpr int NT = 8;
    int wave = (blockIdx.x * blockDim.x + threadIdx.x) >> 5;
    int lane = threadIdx.x & 31;
    int mTile = wave % MT;
    int nBase = (wave / MT) * (16 * NT);
    if (nBase >= NN_) return;
    int g   = lane >> 4;
    int col = lane & 15;   // A row / B column / C column within tile

    // A fragments (weights) — loaded once, held in registers.
    v16h a[KT];
#pragma unroll
    for (int kt = 0; kt < KT; ++kt) {
        const _Float16* wp = W + (mTile * 16 + col) * CIN + kt * 32 + g * 8;
        a[kt] = cat8(*(const v8h*)wp, *(const v8h*)(wp + 16));
    }

    float ps[8], pq[8];
#pragma unroll
    for (int r = 0; r < 8; ++r) { ps[r] = 0.f; pq[r] = 0.f; }

    const _Float16* xrow = X + (size_t)(nBase + col) * CIN + g * 8;

    // prime double buffer with subtile 0
    v8h blo[KT], bhi[KT], nlo[KT], nhi[KT];
#pragma unroll
    for (int kt = 0; kt < KT; ++kt) {
        blo[kt] = *(const v8h*)(xrow + kt * 32);
        bhi[kt] = *(const v8h*)(xrow + kt * 32 + 16);
    }

#pragma unroll
    for (int nt = 0; nt < NT; ++nt) {
        // issue next subtile's fragment loads before this subtile's math
        if (nt + 1 < NT) {
            const _Float16* xn = xrow + (size_t)(nt + 1) * 16 * CIN;
#pragma unroll
            for (int kt = 0; kt < KT; ++kt) {
                nlo[kt] = *(const v8h*)(xn + kt * 32);
                nhi[kt] = *(const v8h*)(xn + kt * 32 + 16);
            }
        }
        if (nt + 2 < NT)
            __builtin_prefetch(xrow + (size_t)(nt + 2) * 16 * CIN, 0, 1);

        v8f c;
#pragma unroll
        for (int r = 0; r < 8; ++r) c[r] = 0.f;
#pragma unroll
        for (int kt = 0; kt < KT; ++kt)
            c = __builtin_amdgcn_wmma_f32_16x16x32_f16(
                    false, a[kt], false, cat8(blo[kt], bhi[kt]), (short)0, c, false, false);

        int n0 = nBase + nt * 16;
        v8h yh;
#pragma unroll
        for (int r = 0; r < 8; ++r) {
            float f = c[r];
            ps[r] += f; pq[r] += f * f;
            yh[r] = (_Float16)f;
        }
        *(v8h*)(Y + (size_t)(n0 + col) * COUT + mTile * 16 + 8 * g) = yh;

#pragma unroll
        for (int kt = 0; kt < KT; ++kt) { blo[kt] = nlo[kt]; bhi[kt] = nhi[kt]; }
    }

    // BN statistics: reduce the 16-sample tile dimension across lanes.
#pragma unroll
    for (int off = 8; off >= 1; off >>= 1) {
#pragma unroll
        for (int r = 0; r < 8; ++r) {
            ps[r] += __shfl_xor(ps[r], off, 32);
            pq[r] += __shfl_xor(pq[r], off, 32);
        }
    }
    if (col == 0) {
        int chBase = mTile * 16 + 8 * g;
#pragma unroll
        for (int r = 0; r < 8; ++r) {
            atomicAdd(&sumv[chBase + r], ps[r]);
            atomicAdd(&sumsq[chBase + r], pq[r]);
        }
    }
}

// ---------------------------------------------------------------------------
// 7. BN finalize: per-channel scale/bias from sum/sumsq
// ---------------------------------------------------------------------------
__global__ void bn_finalize(const float* __restrict__ sumv, const float* __restrict__ sq,
                            const float* __restrict__ gg, const float* __restrict__ bb,
                            float* __restrict__ scale, float* __restrict__ bias,
                            int C, float inv_n) {
    int c = threadIdx.x;
    if (c >= C) return;
    float m = sumv[c] * inv_n;
    float v = sq[c] * inv_n - m * m;
    float sc = gg[c] * rsqrtf(v + BN_EPS_);
    scale[c] = sc;
    bias[c]  = bb[c] - m * sc;
}

// ---------------------------------------------------------------------------
// 8. BN + LeakyReLU in place on f16 activation buffer (v8h chunks)
// ---------------------------------------------------------------------------
__global__ void bn_act(_Float16* __restrict__ X, const float* __restrict__ scale,
                       const float* __restrict__ bias, int CoutPad, int total8) {
    int tid = blockIdx.x * blockDim.x + threadIdx.x;
    if (tid >= total8) return;
    int ch0 = (int)(((long)tid * 8) % CoutPad);
    v8h x = ((v8h*)X)[tid];
#pragma unroll
    for (int r = 0; r < 8; ++r) {
        float f = scale[ch0 + r] * (float)x[r] + bias[ch0 + r];
        f = f > 0.f ? f : LEAKY_ * f;
        x[r] = (_Float16)f;
    }
    ((v8h*)X)[tid] = x;
}

// ---------------------------------------------------------------------------
// 9. final layer: BN + LeakyReLU + max over K=16 neighbors -> feat [B,128,S]
// ---------------------------------------------------------------------------
__global__ void bn_act_max(const _Float16* __restrict__ X3, const float* __restrict__ scale,
                           const float* __restrict__ bias, float* __restrict__ feat) {
    int tid = blockIdx.x * blockDim.x + threadIdx.x;   // B*S*16 (16 = 128ch/8)
    if (tid >= B_ * S_ * 16) return;
    int bs = tid >> 4;
    int c8 = tid & 15;
    const v8h* base = (const v8h*)(X3 + (size_t)bs * 16 * 128) + c8;
    float mx[8];
#pragma unroll
    for (int r = 0; r < 8; ++r) mx[r] = -1e30f;
    for (int k = 0; k < 16; ++k) {
        v8h x = base[(size_t)k * 16];
#pragma unroll
        for (int r = 0; r < 8; ++r) {
            float f = scale[c8 * 8 + r] * (float)x[r] + bias[c8 * 8 + r];
            f = f > 0.f ? f : LEAKY_ * f;
            mx[r] = fmaxf(mx[r], f);
        }
    }
    int b = bs / S_, s = bs % S_;
#pragma unroll
    for (int r = 0; r < 8; ++r)
        feat[((size_t)b * 128 + c8 * 8 + r) * S_ + s] = mx[r];
}

// ---------------------------------------------------------------------------
// Host launcher
// ---------------------------------------------------------------------------
extern "C" void kernel_launch(void* const* d_in, const int* in_sizes, int n_in,
                              void* d_out, int out_size, void* d_ws, size_t ws_size,
                              hipStream_t stream) {
    const float* xyz     = (const float*)d_in[0];
    const float* points  = (const float*)d_in[1];
    const float* conv0_w = (const float*)d_in[2];
    const float* conv1_w = (const float*)d_in[3];
    const float* conv2_w = (const float*)d_in[4];
    const float* bn0_g = (const float*)d_in[5];
    const float* bn0_b = (const float*)d_in[6];
    const float* bn1_g = (const float*)d_in[7];
    const float* bn1_b = (const float*)d_in[8];
    const float* bn2_g = (const float*)d_in[9];
    const float* bn2_b = (const float*)d_in[10];

    char* ws = (char*)d_ws;
    size_t off = 0;
    auto alloc = [&](size_t bytes) -> void* {
        void* p = ws + off;
        off = (off + bytes + 255) & ~(size_t)255;
        return p;
    };

    float*    newxyz_ws = (float*)   alloc((size_t)B_ * S_ * 3 * 4);
    float*    dist_ws   = (float*)   alloc((size_t)B_ * N_ * 4);
    int*      idx_ws    = (int*)     alloc((size_t)B_ * S_ * 16 * 4);
    _Float16* ptsT      = (_Float16*)alloc((size_t)B_ * N_ * 64 * 2);
    float*    stats     = (float*)   alloc(1536 * 4);   // 12 x 128
    _Float16* Wp0       = (_Float16*)alloc((size_t)64 * 96 * 2);
    _Float16* Wp1       = (_Float16*)alloc((size_t)64 * 64 * 2);
    _Float16* Wp2       = (_Float16*)alloc((size_t)128 * 64 * 2);
    _Float16* X0        = (_Float16*)alloc((size_t)NN_ * 96 * 2);
    _Float16* X1        = (_Float16*)alloc((size_t)NN_ * 64 * 2);
    _Float16* X2        = (_Float16*)alloc((size_t)NN_ * 64 * 2);
    _Float16* X3        = (_Float16*)alloc((size_t)NN_ * 128 * 2);

    float* sum0 = stats, *sq0 = stats + 128, *sc0 = stats + 256, *bi0 = stats + 384;
    float* sum1 = stats + 512, *sq1 = stats + 640, *sc1 = stats + 768, *bi1 = stats + 896;
    float* sum2 = stats + 1024, *sq2 = stats + 1152, *sc2 = stats + 1280, *bi2 = stats + 1408;

    float* out_newxyz = (float*)d_out;                        // [B,3,S]
    float* out_feat   = (float*)d_out + (size_t)B_ * 3 * S_;  // [B,128,S]
    int*   out_idx    = (int*)d_out + (size_t)B_ * 3 * S_ + (size_t)B_ * 128 * S_;

    const float inv_n = 1.0f / (float)NN_;

    zero_f32<<<6, 256, 0, stream>>>(stats, 1536);

    pack_w<<<(64 * 96 + 255) / 256, 256, 0, stream>>>(conv0_w, Wp0, 64, 67, 96, 1);
    pack_w<<<(64 * 64 + 255) / 256, 256, 0, stream>>>(conv1_w, Wp1, 64, 64, 64, 0);
    pack_w<<<(128 * 64 + 255) / 256, 256, 0, stream>>>(conv2_w, Wp2, 128, 64, 64, 0);

    pts_transpose<<<(B_ * 64 * N_) / 256, 256, 0, stream>>>(points, ptsT);

    fps_kernel<<<B_, 1024, 0, stream>>>(xyz, dist_ws, newxyz_ws, out_newxyz, out_idx);

    knn_kernel<<<(B_ * S_) / 8, 256, 0, stream>>>(xyz, newxyz_ws, idx_ws);

    gather_kernel<<<NN_ / 256, 256, 0, stream>>>(xyz, ptsT, newxyz_ws, idx_ws, X0);

    // layer 0: 96 -> 64    (mTiles=4, kTiles=3) : waves = 4 * 1024
    gemm_bn<3, 96, 64, 4><<<(4 * (NN_ / 128) * 32) / 256, 256, 0, stream>>>(
        X0, Wp0, X1, sum0, sq0);
    bn_finalize<<<1, 128, 0, stream>>>(sum0, sq0, bn0_g, bn0_b, sc0, bi0, 64, inv_n);
    bn_act<<<(NN_ * 64 / 8) / 256, 256, 0, stream>>>(X1, sc0, bi0, 64, NN_ * 64 / 8);

    // layer 1: 64 -> 64
    gemm_bn<2, 64, 64, 4><<<(4 * (NN_ / 128) * 32) / 256, 256, 0, stream>>>(
        X1, Wp1, X2, sum1, sq1);
    bn_finalize<<<1, 128, 0, stream>>>(sum1, sq1, bn1_g, bn1_b, sc1, bi1, 64, inv_n);
    bn_act<<<(NN_ * 64 / 8) / 256, 256, 0, stream>>>(X2, sc1, bi1, 64, NN_ * 64 / 8);

    // layer 2: 64 -> 128   (mTiles=8)
    gemm_bn<2, 64, 128, 8><<<(8 * (NN_ / 128) * 32) / 256, 256, 0, stream>>>(
        X2, Wp2, X3, sum2, sq2);
    bn_finalize<<<1, 128, 0, stream>>>(sum2, sq2, bn2_g, bn2_b, sc2, bi2, 128, inv_n);

    bn_act_max<<<(B_ * S_ * 16) / 256, 256, 0, stream>>>(X3, sc2, bi2, out_feat);
}
